// Transformer_36009005810066
// MI455X (gfx1250) — matrix operand order repaired and checked by reading, not compiled, not run on hardware
//
#include <hip/hip_runtime.h>

// ---------------------------------------------------------------------------
// Transformer block for MI455X (gfx1250): bf16 WMMA GEMMs + f32 accumulate.
// B=2, P=4096, E=512, H=8, D=64. All GEMM dims divide the 128x64 block tile.
// V is stored transposed [B,H,D,P] so every GEMM operand loads as b128.
// ---------------------------------------------------------------------------

#define DEV static __device__ __forceinline__

constexpr int Bb = 2;
constexpr int Pp = 4096;
constexpr int Ee = 512;
constexpr int Hh = 8;
constexpr int Dd = 64;
constexpr int PE = Pp * Ee;          // 2^21
constexpr int Mrows = Bb * Pp;       // 8192
constexpr int N3 = 3 * Ee;           // 1536
constexpr long long PD = (long long)Pp * Dd; // 262144 (per-head elems, same for v^T)

typedef __attribute__((ext_vector_type(16))) __bf16 bf16x16;
typedef __attribute__((ext_vector_type(8)))  float  f32x8;
typedef __attribute__((ext_vector_type(4)))  unsigned int u32x4;

union Frag {
  unsigned short u[16];
  u32x4          q[2];
  bf16x16        v;
};

DEV unsigned short f32_to_bf16(float f) {
  unsigned int u = __float_as_uint(f);
  unsigned int r = (u + 0x7FFFu + ((u >> 16) & 1u)) >> 16;   // RNE
  return (unsigned short)r;
}
DEV float geluf(float x) { return 0.5f * x * (1.0f + erff(x * 0.70710678118654752f)); }

// ---- WMMA fragment loaders (layouts per CDNA5 ISA 7.12.2) ------------------
// A-matrix 16x32 bf16: lane half hf in {0,1}, row = idx; element e -> K:
//   e<8 : hf*8 + e            (two contiguous 16B chunks)
//   e>=8: 16 + hf*8 + (e-8)
DEV void load_fragA(const unsigned short* base, int ld, int row, int kb, int hf, Frag& f) {
  const unsigned short* p = base + (size_t)row * ld + kb + hf * 8;
  f.q[0] = *reinterpret_cast<const u32x4*>(p);
  f.q[1] = *reinterpret_cast<const u32x4*>(p + 16);
}
// B-matrix 32x16 bf16: lane idx = column n; element e -> K same as A (e<8: hf*8+e, e>=8: 16+hf*8+(e-8)).
DEV void load_fragB(const unsigned short* base, int ld, int col, int kb, int hf, Frag& f) {
  const unsigned short* p = base + (size_t)col * ld + kb + hf * 8;
  f.q[0] = *reinterpret_cast<const u32x4*>(p);
  f.q[1] = *reinterpret_cast<const u32x4*>(p + 16);
}

DEV f32x8 wmma_bf16(const Frag& a, const Frag& b, f32x8 c) {
  return __builtin_amdgcn_wmma_f32_16x16x32_bf16(false, a.v, false, b.v, (short)0, c,
                                                 false, false);
}

// Per-wave 32x32 tile: 2x2 accumulators, 4 WMMAs per K-step.
// Both operands K-contiguous: A row-major [M,K] (lda), B row-major [N,K] (ldb).
DEV void gemm_core(const unsigned short* A, int lda, const unsigned short* Bm, int ldb,
                   int K, int m0, int n0, int lane, f32x8 acc[2][2]) {
  const int hf = lane >> 4, idx = lane & 15;
  for (int kb = 0; kb < K; kb += 32) {
    Frag a0, a1, b0, b1;
    load_fragA(A, lda, m0 + idx,      kb, hf, a0);
    load_fragA(A, lda, m0 + 16 + idx, kb, hf, a1);
    load_fragB(Bm, ldb, n0 + idx,      kb, hf, b0);
    load_fragB(Bm, ldb, n0 + 16 + idx, kb, hf, b1);
    acc[0][0] = wmma_bf16(a0, b0, acc[0][0]);
    acc[0][1] = wmma_bf16(a0, b1, acc[0][1]);
    acc[1][0] = wmma_bf16(a1, b0, acc[1][0]);
    acc[1][1] = wmma_bf16(a1, b1, acc[1][1]);
  }
}

// C element (within 16x16 tile): m = (lane>=16 ? 8 : 0) + r, n = lane & 15.
// Variadic so commas inside the body don't split macro arguments.
#define EPILOGUE_LOOP(...)                                           \
  {                                                                  \
    const int lhalf = (lane >> 4) * 8, lcol = lane & 15;             \
    _Pragma("unroll") for (int tm = 0; tm < 2; ++tm)                 \
    _Pragma("unroll") for (int tn = 0; tn < 2; ++tn)                 \
    _Pragma("unroll") for (int r = 0; r < 8; ++r) {                  \
      const int m = m0 + tm * 16 + lhalf + r;                        \
      const int n = n0 + tn * 16 + lcol;                             \
      const float cval = acc[tm][tn][r];                             \
      __VA_ARGS__                                                    \
    }                                                                \
  }

#define GEMM_PROLOGUE()                                              \
  const int lane = threadIdx.x & 31;                                 \
  const int wave = threadIdx.x >> 5;                                 \
  const int m0 = blockIdx.x * 128 + (wave >> 1) * 32;                \
  const int n0 = blockIdx.y * 64 + (wave & 1) * 32;                  \
  f32x8 acc[2][2] = {};

// ---------------------------------------------------------------------------
// Elementwise / reduction kernels
// ---------------------------------------------------------------------------
__global__ void k_f32_to_bf16(const float* __restrict__ src, unsigned short* __restrict__ dst,
                              int n) {
  for (int i = blockIdx.x * blockDim.x + threadIdx.x; i < n; i += gridDim.x * blockDim.x)
    dst[i] = f32_to_bf16(src[i]);
}

// Stage 1: 512 blocks (256 per batch), each reduces 8192 contiguous floats.
__global__ void k_ln_part(const float* __restrict__ x, float2* __restrict__ partial) {
  const int b = blockIdx.x >> 8, blk = blockIdx.x & 255;
  const float* p = x + (size_t)b * PE + blk * 8192 + threadIdx.x;
  float s = 0.f, q = 0.f;
#pragma unroll 4
  for (int i = 0; i < 32; ++i) { float v = p[i * 256]; s += v; q += v * v; }
  __shared__ float ss[256], sq[256];
  ss[threadIdx.x] = s; sq[threadIdx.x] = q; __syncthreads();
  for (int st = 128; st > 0; st >>= 1) {
    if (threadIdx.x < st) { ss[threadIdx.x] += ss[threadIdx.x + st]; sq[threadIdx.x] += sq[threadIdx.x + st]; }
    __syncthreads();
  }
  if (threadIdx.x == 0) partial[blockIdx.x] = make_float2(ss[0], sq[0]);
}

__global__ void k_ln_stats(const float2* __restrict__ partial, float2* __restrict__ stats) {
  const int b = blockIdx.x;
  __shared__ float ss[256], sq[256];
  float2 v = partial[b * 256 + threadIdx.x];
  ss[threadIdx.x] = v.x; sq[threadIdx.x] = v.y; __syncthreads();
  for (int st = 128; st > 0; st >>= 1) {
    if (threadIdx.x < st) { ss[threadIdx.x] += ss[threadIdx.x + st]; sq[threadIdx.x] += sq[threadIdx.x + st]; }
    __syncthreads();
  }
  if (threadIdx.x == 0) {
    const float inv = 1.0f / (float)PE;
    float mean = ss[0] * inv;
    float var  = sq[0] * inv - mean * mean;
    stats[b] = make_float2(mean, rsqrtf(var + 1e-5f));
  }
}

__global__ void k_ln_apply(const float* __restrict__ x, const float* __restrict__ g,
                           const float* __restrict__ bia, const float2* __restrict__ stats,
                           unsigned short* __restrict__ out, int ntot) {
  for (int i = blockIdx.x * blockDim.x + threadIdx.x; i < ntot; i += gridDim.x * blockDim.x) {
    const int b = i >> 21, pe = i & (PE - 1);
    const float2 st = stats[b];
    out[i] = f32_to_bf16((x[i] - st.x) * st.y * g[pe] + bia[pe]);
  }
}

// Column softmax (axis = query). Thread t owns column k; adjacent threads ->
// adjacent addresses, so every pass is coalesced.
__global__ void k_softmax_col(const float* __restrict__ S, unsigned short* __restrict__ dpb) {
  const int col = blockIdx.x * 256 + threadIdx.x;
  float mx = -3.402823466e38f;
  for (int q = 0; q < Pp; ++q) mx = fmaxf(mx, S[(size_t)q * Pp + col]);
  float s = 0.f;
  for (int q = 0; q < Pp; ++q) s += expf(S[(size_t)q * Pp + col] - mx);
  const float inv = 1.0f / s;
  for (int q = 0; q < Pp; ++q)
    dpb[(size_t)q * Pp + col] = f32_to_bf16(expf(S[(size_t)q * Pp + col] - mx) * inv);
}

// ---------------------------------------------------------------------------
// GEMM kernels (WMMA bf16 -> f32)
// ---------------------------------------------------------------------------
// qkv = hn[8192,512] @ W1[1536,512]^T.
// q,k stored [B,H,P,D]; v stored TRANSPOSED [B,H,D,P] so the P.V GEMM's B
// operand is K-contiguous (b128 loads instead of 16 scalar u16 gathers).
__global__ void k_gemm_qkv(const unsigned short* __restrict__ A,
                           const unsigned short* __restrict__ W,
                           unsigned short* __restrict__ qkv) {
  GEMM_PROLOGUE()
  gemm_core(A, Ee, W, Ee, Ee, m0, n0, lane, acc);
  EPILOGUE_LOOP(
    const int b = m >> 12;
    const int p = m & (Pp - 1);
    const int three = n >> 9;
    const int h = (n >> 6) & 7;
    const int d = n & 63;
    const size_t head = ((size_t)three * Bb + b) * Hh + h;
    const size_t idx = (three == 2) ? (head * Dd + d) * Pp + p    // v^T [D,P]
                                    : (head * Pp + p) * Dd + d;   // q,k [P,D]
    qkv[idx] = f32_to_bf16(cval);
  )
}

// S = q[P,D] @ k[P,D]^T * 1/sqrt(D)   (one head per launch)
__global__ void k_gemm_scores(const unsigned short* __restrict__ q,
                              const unsigned short* __restrict__ kk,
                              float* __restrict__ S) {
  GEMM_PROLOGUE()
  gemm_core(q, Dd, kk, Dd, Dd, m0, n0, lane, acc);
  EPILOGUE_LOOP(
    S[(size_t)m * Pp + n] = cval * 0.125f;
  )
}

// wv = dp[P,P] @ v[P,D]; with v^T [D,P] the B operand is K-contiguous (ldb=P).
// out1 = x + wv (heads write disjoint E-columns).
__global__ void k_gemm_wv(const unsigned short* __restrict__ dp,
                          const unsigned short* __restrict__ vT,
                          const float* __restrict__ x, float* __restrict__ out1,
                          int b, int h) {
  GEMM_PROLOGUE()
  gemm_core(dp, Pp, vT, Pp, Pp, m0, n0, lane, acc);
  EPILOGUE_LOOP(
    const size_t idx = ((size_t)b * Pp + m) * Ee + h * Dd + n;
    out1[idx] = x[idx] + cval;
  )
}

// t1 = gelu(h2n[8192,512] @ Wa[1536,512]^T + ba)  (bf16 out)
__global__ void k_gemm_mlp1(const unsigned short* __restrict__ A,
                            const unsigned short* __restrict__ W,
                            const float* __restrict__ bias,
                            unsigned short* __restrict__ t1) {
  GEMM_PROLOGUE()
  gemm_core(A, Ee, W, Ee, Ee, m0, n0, lane, acc);
  EPILOGUE_LOOP(
    t1[(size_t)m * N3 + n] = f32_to_bf16(geluf(cval + bias[n]));
  )
}

// out = gelu(t1[8192,1536] @ Wb[512,1536]^T + bb) + out1   (f32 out)
__global__ void k_gemm_mlp2(const unsigned short* __restrict__ A,
                            const unsigned short* __restrict__ W,
                            const float* __restrict__ bias,
                            const float* __restrict__ out1, float* __restrict__ out) {
  GEMM_PROLOGUE()
  gemm_core(A, N3, W, N3, N3, m0, n0, lane, acc);
  EPILOGUE_LOOP(
    const size_t idx = (size_t)m * Ee + n;
    out[idx] = geluf(cval + bias[n]) + out1[idx];
  )
}

// ---------------------------------------------------------------------------
extern "C" void kernel_launch(void* const* d_in, const int* in_sizes, int n_in,
                              void* d_out, int out_size, void* d_ws, size_t ws_size,
                              hipStream_t stream) {
  const float* x  = (const float*)d_in[0];
  const float* g1 = (const float*)d_in[1];
  const float* b1 = (const float*)d_in[2];
  const float* W1 = (const float*)d_in[3];
  const float* g2 = (const float*)d_in[4];
  const float* b2 = (const float*)d_in[5];
  const float* Wa = (const float*)d_in[6];
  const float* ba = (const float*)d_in[7];
  const float* Wb = (const float*)d_in[8];
  const float* bbv = (const float*)d_in[9];
  float* out = (float*)d_out;

  char* w = (char*)d_ws;
  size_t cur = 0;
  auto alloc = [&](size_t bytes) -> void* {
    void* p = (void*)(w + cur);
    cur += (bytes + 255) & ~(size_t)255;
    return p;
  };
  float2* partial = (float2*)alloc((size_t)Bb * 256 * sizeof(float2));
  float2* stats1  = (float2*)alloc(Bb * sizeof(float2));
  float2* stats2  = (float2*)alloc(Bb * sizeof(float2));
  unsigned short* W1b  = (unsigned short*)alloc((size_t)N3 * Ee * 2);
  unsigned short* Wab  = (unsigned short*)alloc((size_t)N3 * Ee * 2);
  unsigned short* Wbb  = (unsigned short*)alloc((size_t)Ee * N3 * 2);
  unsigned short* hn   = (unsigned short*)alloc((size_t)Mrows * Ee * 2);   // LN1 out; reused for LN2 out
  unsigned short* qkvb = (unsigned short*)alloc((size_t)3 * Bb * Hh * Pp * Dd * 2);
  float*          S    = (float*)alloc((size_t)Pp * Pp * 4);               // reused as t1 (bf16) later
  unsigned short* dpb  = (unsigned short*)alloc((size_t)Pp * Pp * 2);
  float*          out1 = (float*)alloc((size_t)Mrows * Ee * 4);
  unsigned short* t1   = (unsigned short*)S;                               // 24 MB inside 64 MB region
  (void)ws_size; (void)in_sizes; (void)n_in; (void)out_size;

  const int nW = N3 * Ee;  // 786432 (same for W1, Wa, Wb)
  k_f32_to_bf16<<<1024, 256, 0, stream>>>(W1, W1b, nW);
  k_f32_to_bf16<<<1024, 256, 0, stream>>>(Wa, Wab, nW);
  k_f32_to_bf16<<<1024, 256, 0, stream>>>(Wb, Wbb, nW);

  // ---- LayerNorm 1 (over [P,E] jointly per batch) ----
  k_ln_part<<<Bb * 256, 256, 0, stream>>>(x, partial);
  k_ln_stats<<<Bb, 256, 0, stream>>>(partial, stats1);
  k_ln_apply<<<4096, 256, 0, stream>>>(x, g1, b1, stats1, hn, Bb * PE);

  // ---- QKV projection ----
  k_gemm_qkv<<<dim3(Mrows / 128, N3 / 64), 256, 0, stream>>>(hn, W1b, qkvb);

  // ---- Attention, one (b,h) head at a time (scores buffer reused) ----
  for (int b = 0; b < Bb; ++b) {
    for (int h = 0; h < Hh; ++h) {
      const unsigned short* qp = qkvb + (size_t)((0 * Bb + b) * Hh + h) * PD;
      const unsigned short* kp = qkvb + (size_t)((1 * Bb + b) * Hh + h) * PD;
      const unsigned short* vp = qkvb + (size_t)((2 * Bb + b) * Hh + h) * PD; // v^T [D,P]
      k_gemm_scores<<<dim3(Pp / 128, Pp / 64), 256, 0, stream>>>(qp, kp, S);
      k_softmax_col<<<Pp / 256, 256, 0, stream>>>(S, dpb);
      k_gemm_wv<<<dim3(Pp / 128, Dd / 64), 256, 0, stream>>>(dpb, vp, x, out1, b, h);
    }
  }

  // ---- LayerNorm 2 on out1 ----
  k_ln_part<<<Bb * 256, 256, 0, stream>>>(out1, partial);
  k_ln_stats<<<Bb, 256, 0, stream>>>(partial, stats2);
  k_ln_apply<<<4096, 256, 0, stream>>>(out1, g2, b2, stats2, hn, Bb * PE);

  // ---- MLP ----
  k_gemm_mlp1<<<dim3(Mrows / 128, N3 / 64), 256, 0, stream>>>(hn, Wab, ba, t1);
  k_gemm_mlp2<<<dim3(Mrows / 128, Ee / 64), 256, 0, stream>>>(t1, Wbb, bbv, out1, out);
}
